// UResNet_38319698215235
// MI455X (gfx1250) — matrix operand order repaired and checked by reading, not compiled, not run on hardware
//
#include <hip/hip_runtime.h>
#include <math.h>

// ---------------------------------------------------------------------------
// Sparse 3D UResNet forward for MI455X (gfx1250, wave32, WMMA).
// Core op: rulebook gather-GEMM-scatter with v_wmma_f32_16x16x32_bf16.
// Kernels are templated on (CinP, Cout) — every shape in this net is static —
// so N/K loops fully unroll: A fragments are gathered once per 16-pair tile
// (two global_load_b128 per K-step, bf16-packed, 16B-aligned via stride
// padding to multiples of 8) and reused across all Cout/16 N-tiles.
// f32 accumulate, hardware f32 atomics for scatter. Working set is
// L2-resident (192 MB); bf16 operands halve the dominant gather traffic.
// ---------------------------------------------------------------------------

typedef __attribute__((ext_vector_type(16))) __bf16 v16bf;
typedef __attribute__((ext_vector_type(8)))  float  v8f;
typedef __attribute__((ext_vector_type(8)))  int    v8i_t;
typedef __attribute__((ext_vector_type(4)))  int    v4i_t;

#define TPB 256
#define WPB 8   // waves per block (wave32)

__device__ __forceinline__ v4i_t ld128(const __bf16* p) { return *(const v4i_t*)p; }

// ---- sparse conv: out[dst] += x_bf16[src] @ Wt  (f32 accum, f32 atomics) ---
// x  : bf16 activations, row stride CINP (multiple of 8, zero-padded)
// Wt : bf16 weights for ONE offset, layout [COUT][CINP] (K-contiguous)
template<int CINP, int COUT>
__global__ __launch_bounds__(TPB) void spconv_wmma_t(
    const __bf16* __restrict__ x, const __bf16* __restrict__ Wt,
    const int2* __restrict__ pairs, int npairs,
    float* __restrict__ y, int transpose)
{
    constexpr int NK = (CINP + 31) / 32;
    constexpr int NN = COUT / 16;            // conv Cout is a multiple of 16

    const int lane = threadIdx.x & 31;
    const int wave = threadIdx.x >> 5;
    const int base = (blockIdx.x * WPB + wave) * 16;
    if (base >= npairs) return;              // wave-uniform exit (EXEC all-1)

    const int h    = (lane >> 4) & 1;        // half: lanes 0-15 / 16-31
    const int mrow = lane & 15;              // A row owned by this lane
    const int ncol = lane & 15;              // C/D column owned by this lane

    // tail rows read row 0 (C rows are independent; tails never scattered)
    int src = 0;
    if (base + mrow < npairs) {
        int2 pr = pairs[base + mrow];
        src = transpose ? pr.y : pr.x;
    }
    const __bf16* __restrict__ xrow = x + (size_t)src * CINP;

    int dsts[8];
#pragma unroll
    for (int r = 0; r < 8; ++r) {
        int m = r + (h << 3);
        dsts[r] = -1;
        if (base + m < npairs) {
            int2 pr = pairs[base + m];
            dsts[r] = transpose ? pr.x : pr.y;
        }
    }

    // prefetch next tile's pair list (global_prefetch_b8)
    if (base + 16 + mrow < npairs)
        __builtin_prefetch(&pairs[base + 16 + mrow], 0, 1);

    // ---- gather ALL A fragments once; reuse across the N-tiles ----
    // A 16x32 bf16 layout (ISA 7.12.2): halfs e=0..7 -> K=ko+8h.., e=8..15 ->
    // K=ko+16+8h.. : two contiguous 8-element runs = two b128 loads.
    v8i_t afrag[NK];
#pragma unroll
    for (int kt = 0; kt < NK; ++kt) {
        int ko = kt * 32;
        v4i_t lo = {}, hi = {};
        int s0 = ko + (h << 3);
        int s1 = ko + 16 + (h << 3);
        if (s0 < CINP) lo = ld128(xrow + s0);
        if (s1 < CINP) hi = ld128(xrow + s1);
        afrag[kt] = __builtin_shufflevector(lo, hi, 0, 1, 2, 3, 4, 5, 6, 7);
    }

    v8f acc[NN];
#pragma unroll
    for (int nt = 0; nt < NN; ++nt) acc[nt] = {};

#pragma unroll
    for (int nt = 0; nt < NN; ++nt) {
        const __bf16* __restrict__ wrow = Wt + (size_t)(nt * 16 + ncol) * CINP;
#pragma unroll
        for (int kt = 0; kt < NK; ++kt) {
            int ko = kt * 32;
            // B 32x16: halfs e=0..7 -> K=ko+8h.., e=8..15 -> K=ko+16+8h.. (matches A).
            v4i_t lo = {}, hi = {};
            int s0 = ko + (h << 3);
            int s1 = ko + 16 + (h << 3);
            if (s0 < CINP) lo = ld128(wrow + s0);
            if (s1 < CINP) hi = ld128(wrow + s1);
            v8i_t bi = __builtin_shufflevector(lo, hi, 0, 1, 2, 3, 4, 5, 6, 7);
            acc[nt] = __builtin_amdgcn_wmma_f32_16x16x32_bf16(
                          false, __builtin_bit_cast(v16bf, afrag[kt]),
                          false, __builtin_bit_cast(v16bf, bi),
                          (short)0, acc[nt], false, false);
        }
    }

#pragma unroll
    for (int nt = 0; nt < NN; ++nt)
#pragma unroll
        for (int r = 0; r < 8; ++r)
            if (dsts[r] >= 0)
                unsafeAtomicAdd(&y[(size_t)dsts[r] * COUT + nt * 16 + ncol],
                                acc[nt][r]);
}

// ---- dense GEMM (NiN shortcuts, heads): y = x @ W (+bias), plain stores ----
template<int CINP, int COUT>
__global__ __launch_bounds__(TPB) void dense_wmma_t(
    const __bf16* __restrict__ x, const __bf16* __restrict__ Wt,
    const float* __restrict__ bias, float* __restrict__ y,
    const int* __restrict__ n_dev)
{
    constexpr int NK = (CINP + 31) / 32;
    constexpr int NN = (COUT + 15) / 16;

    const int lane = threadIdx.x & 31;
    const int wave = threadIdx.x >> 5;
    const int base = (blockIdx.x * WPB + wave) * 16;
    const int n    = *n_dev;
    if (base >= n) return;                    // wave-uniform

    const int h    = (lane >> 4) & 1;
    const int mrow = lane & 15;
    const int ncol = lane & 15;
    const int row  = base + mrow;
    const __bf16* __restrict__ xrow = x + (size_t)(row < n ? row : 0) * CINP;

    v8i_t afrag[NK];
#pragma unroll
    for (int kt = 0; kt < NK; ++kt) {
        int ko = kt * 32;
        v4i_t lo = {}, hi = {};
        int s0 = ko + (h << 3);
        int s1 = ko + 16 + (h << 3);
        if (s0 < CINP) lo = ld128(xrow + s0);
        if (s1 < CINP) hi = ld128(xrow + s1);
        afrag[kt] = __builtin_shufflevector(lo, hi, 0, 1, 2, 3, 4, 5, 6, 7);
    }

    v8f acc[NN];
#pragma unroll
    for (int nt = 0; nt < NN; ++nt) acc[nt] = {};

#pragma unroll
    for (int nt = 0; nt < NN; ++nt) {
        int nc = nt * 16 + ncol;
        const __bf16* __restrict__ wrow =
            Wt + (size_t)(nc < COUT ? nc : COUT - 1) * CINP;   // clamp; stores guarded
#pragma unroll
        for (int kt = 0; kt < NK; ++kt) {
            int ko = kt * 32;
            v4i_t lo = {}, hi = {};
            int s0 = ko + (h << 3);
            int s1 = ko + 16 + (h << 3);
            if (s0 < CINP) lo = ld128(wrow + s0);
            if (s1 < CINP) hi = ld128(wrow + s1);
            v8i_t bi = __builtin_shufflevector(lo, hi, 0, 1, 2, 3, 4, 5, 6, 7);
            acc[nt] = __builtin_amdgcn_wmma_f32_16x16x32_bf16(
                          false, __builtin_bit_cast(v16bf, afrag[kt]),
                          false, __builtin_bit_cast(v16bf, bi),
                          (short)0, acc[nt], false, false);
        }
    }

#pragma unroll
    for (int nt = 0; nt < NN; ++nt) {
        int nc = nt * 16 + ncol;
#pragma unroll
        for (int r = 0; r < 8; ++r) {
            int orow = base + r + (h << 3);
            if (orow < n && nc < COUT)
                y[(size_t)orow * COUT + nc] = acc[nt][r] + (bias ? bias[nc] : 0.0f);
        }
    }
}

// ---- weight prep: f32 [K][Cin][Cout] -> bf16 [K][Cout][CinP], zero-pad -----
__global__ void wprep_kernel(const float* __restrict__ W, __bf16* __restrict__ Wt,
                             int K, int Cin, int Cout, int CinP, int total)
{
    int idx = blockIdx.x * blockDim.x + threadIdx.x;
    if (idx >= total) return;                 // total = K*Cout*CinP (host-known)
    int k   = idx / (Cout * CinP);
    int rem = idx % (Cout * CinP);
    int nn  = rem / CinP;
    int c   = rem % CinP;
    float v = (c < Cin) ? W[((size_t)k * Cin + c) * Cout + nn] : 0.0f;
    Wt[idx] = (__bf16)v;
}

// ---- activation convert: f32 [n][Csrc] -> bf16 [n][Cdst], zero-pad ---------
__global__ void cvt_bf16_kernel(const float* __restrict__ x, __bf16* __restrict__ y,
                                const int* __restrict__ n_dev, int Csrc, int Cdst)
{
    const int n = *n_dev;
    long long idx = (long long)blockIdx.x * blockDim.x + threadIdx.x;
    if (idx >= (long long)n * Cdst) return;
    int r = (int)(idx / Cdst), c = (int)(idx % Cdst);
    y[idx] = (__bf16)(c < Csrc ? x[(size_t)r * Csrc + c] : 0.0f);
}

// ---- per-channel sum / sumsq (LDS ds_add_f32 then global atomics) ----------
__global__ void colstats_kernel(const float* __restrict__ x, float* __restrict__ stats,
                                const int* __restrict__ n_dev, int C)
{
    extern __shared__ float ls[];             // 2*C floats
    for (int i = threadIdx.x; i < 2 * C; i += blockDim.x) ls[i] = 0.0f;
    __syncthreads();
    const int n = *n_dev;
    const long long total  = (long long)n * C;
    const long long stride = (long long)gridDim.x * blockDim.x;
    for (long long idx = (long long)blockIdx.x * blockDim.x + threadIdx.x;
         idx < total; idx += stride) {
        float v = x[idx];
        int   c = (int)(idx % C);
        atomicAdd(&ls[c], v);
        atomicAdd(&ls[C + c], v * v);
    }
    __syncthreads();
    for (int i = threadIdx.x; i < 2 * C; i += blockDim.x)
        unsafeAtomicAdd(&stats[i], ls[i]);
}

// ---- fused batchnorm (biased var) + leaky relu, emits bf16 -----------------
// (every BN output in this net is consumed only by a WMMA kernel)
__global__ void bnact_bf16_kernel(const float* __restrict__ x, __bf16* __restrict__ y,
                                  const float* __restrict__ stats,
                                  const float* __restrict__ g, const float* __restrict__ b,
                                  const int* __restrict__ n_dev, int C, float leak)
{
    const int n = *n_dev;
    long long idx = (long long)blockIdx.x * blockDim.x + threadIdx.x;
    if (idx >= (long long)n * C) return;
    int   c   = (int)(idx % C);
    float inv = 1.0f / (float)n;
    float mu  = stats[c] * inv;
    float var = stats[C + c] * inv - mu * mu;
    float v   = (x[idx] - mu) * rsqrtf(var + 1e-4f) * g[c] + b[c];
    v = v > 0.0f ? v : leak * v;
    y[idx] = (__bf16)v;
}

__global__ void add_kernel(float* __restrict__ y, const float* __restrict__ a,
                           const int* __restrict__ n_dev, int C)
{
    const int n = *n_dev;
    long long idx = (long long)blockIdx.x * blockDim.x + threadIdx.x;
    if (idx >= (long long)n * C) return;
    y[idx] += a[idx];
}

__global__ void copycols_kernel(const float* __restrict__ src, int Csrc,
                                float* __restrict__ dst, int Cdst, int ofs,
                                const int* __restrict__ n_dev)
{
    const int n = *n_dev;
    long long idx = (long long)blockIdx.x * blockDim.x + threadIdx.x;
    if (idx >= (long long)n * Csrc) return;
    int r = (int)(idx / Csrc), c = (int)(idx % Csrc);
    dst[(size_t)r * Cdst + ofs + c] = src[idx];
}

__global__ void sigmoid_col0_kernel(float* __restrict__ y, int C,
                                    const int* __restrict__ n_dev)
{
    const int n = *n_dev;
    int r = blockIdx.x * blockDim.x + threadIdx.x;
    if (r >= n) return;
    float v = y[(size_t)r * C];
    y[(size_t)r * C] = 1.0f / (1.0f + expf(-v));
}

// ===========================================================================
extern "C" void kernel_launch(void* const* d_in, const int* in_sizes, int n_in,
                              void* d_out, int out_size, void* d_ws, size_t ws_size,
                              hipStream_t stream)
{
    (void)n_in; (void)ws_size;
    static const int NPLh[5] = {16, 32, 48, 64, 80};

    // ---- parse inputs (setup_inputs insertion order, depth-first) ----------
    int cur = 0;
    auto F = [&]() -> const float* { return (const float*)d_in[cur++]; };
    auto I = [&]() -> const int*   { return (const int*)d_in[cur++]; };

    const float* features = F();                                   // [n0, 1]

    struct BN  { const float* g; const float* b; };
    struct Blk { BN bn1; const float* w1; BN bn2; const float* w2; const float* nin; };
    auto rdBN  = [&](BN& bn) { bn.g = F(); bn.b = F(); };
    auto rdBlk = [&](Blk& b, bool nin) {
        rdBN(b.bn1); b.w1 = F(); rdBN(b.bn2); b.w2 = F();
        b.nin = nin ? F() : nullptr;
    };

    const float* w_in = F();                                       // [27,1,16]
    Blk enc[5][2];
    for (int i = 0; i < 5; i++) for (int j = 0; j < 2; j++) rdBlk(enc[i][j], false);
    BN down_bn[4]; const float* down_w[4];
    for (int i = 0; i < 4; i++) { rdBN(down_bn[i]); down_w[i] = F(); }
    struct Dec { BN bn; const float* wdec; Blk b0, b1; };
    Dec dec1[4], dec2[4];
    auto rdDec = [&](Dec* d) {
        for (int i = 0; i < 4; i++) {
            rdBN(d[i].bn); d[i].wdec = F();
            rdBlk(d[i].b0, true); rdBlk(d[i].b1, false);
        }
    };
    rdDec(dec1); rdDec(dec2);
    BN seg_bn;  rdBN(seg_bn);  const float* seg_w  = F(); const float* seg_b  = F();
    Blk embB[3];  for (int i = 0; i < 3; i++) rdBlk(embB[i], false);
    BN emb_bn;  rdBN(emb_bn);  const float* emb_w  = F(); const float* emb_b  = F();
    Blk seedB[3]; for (int i = 0; i < 3; i++) rdBlk(seedB[i], false);
    BN seed_bn; rdBN(seed_bn); const float* seed_w = F(); const float* seed_b = F();

    struct RB { const int2* pairs; int np; };
    RB rb_sub[5][27];
    for (int l = 0; l < 5; l++)
        for (int k = 0; k < 27; k++) {
            rb_sub[l][k].pairs = (const int2*)d_in[cur];
            rb_sub[l][k].np    = in_sizes[cur] / 2; cur++;
        }
    RB rb_down[4][8];
    for (int l = 0; l < 4; l++)
        for (int k = 0; k < 8; k++) {
            rb_down[l][k].pairs = (const int2*)d_in[cur];
            rb_down[l][k].np    = in_sizes[cur] / 2; cur++;
        }
    const int* ndev[5];
    for (int l = 0; l < 5; l++) ndev[l] = I();     // per-level row counts (device)

    const int nmax = in_sizes[0];                   // n_levels[0] upper-bounds all
    const int n0   = out_size / 23;                 // seg 5 + emb 16 + seed 2

    // ---- workspace bump allocator (byte-based) -----------------------------
    size_t off = 0;
    auto allocB = [&](size_t bytes) -> void* {
        void* p = (char*)d_ws + off;
        off += (bytes + 255) & ~(size_t)255;
        return p;
    };
    auto allocF = [&](size_t nflt) -> float* { return (float*)allocB(nflt * 4); };
    float*  G0 = allocF((size_t)nmax * 160);
    float*  G1 = allocF((size_t)nmax * 160);
    float*  T2 = allocF((size_t)nmax * 80);
    float*  SCb = allocF((size_t)nmax * 80);
    float*  FM[4]; for (int l = 0; l < 4; l++) FM[l] = allocF((size_t)nmax * NPLh[l]);
    float*  E4 = allocF((size_t)nmax * 80);
    float*  X1 = allocF((size_t)nmax * 16);
    float*  X2 = allocF((size_t)nmax * 16);
    float*  stats = allocF(512);
    __bf16* XH = (__bf16*)allocB((size_t)nmax * 160 * 2);    // packed activations
    __bf16* WT = (__bf16*)allocB((size_t)27 * 80 * 160 * 2); // packed weights

    auto zero = [&](float* p, size_t nf) { hipMemsetAsync(p, 0, nf * sizeof(float), stream); };
    auto d2d  = [&](float* d, const float* s, size_t nf) {
        hipMemcpyAsync(d, s, nf * sizeof(float), hipMemcpyDeviceToDevice, stream);
    };
    auto eb = [](long long t) -> int { return (int)((t + 255) / 256); };

    // ---- static-shape dispatch for the WMMA kernels ------------------------
    auto sp = [&](int CinP, int Cout, const __bf16* xh, const __bf16* Wt,
                  const int2* pairs, int np, float* y, int tr) {
        int g = (((np + 15) / 16) + WPB - 1) / WPB;
#define SPC(CP, CO) else if (CinP == CP && Cout == CO) \
        spconv_wmma_t<CP, CO><<<g, TPB, 0, stream>>>(xh, Wt, pairs, np, y, tr)
        if (false) {}
        SPC(8, 16);  SPC(16, 16); SPC(16, 32); SPC(32, 16); SPC(32, 32);
        SPC(32, 48); SPC(48, 32); SPC(48, 48); SPC(48, 64); SPC(64, 32);
        SPC(64, 48); SPC(64, 64); SPC(64, 80); SPC(80, 64); SPC(80, 80);
        SPC(96, 48); SPC(128, 64);
#undef SPC
    };
    auto dn = [&](int CinP, int Cout, const __bf16* xh, const float* bias,
                  float* y, const int* nd) {
        int g = (((nmax + 15) / 16) + WPB - 1) / WPB;
#define DNC(CP, CO) else if (CinP == CP && Cout == CO) \
        dense_wmma_t<CP, CO><<<g, TPB, 0, stream>>>(xh, WT, bias, y, nd)
        if (false) {}
        DNC(16, 5);  DNC(16, 2);  DNC(16, 16); DNC(32, 16);
        DNC(64, 32); DNC(96, 48); DNC(128, 64);
#undef DNC
    };

    // sparse conv: prep weights once, then one launch per rulebook offset
    auto conv = [&](const __bf16* xh, int CinP, const float* W, const RB* rb, int K,
                    int Cin, int Cout, float* y, bool tr) {
        int wtot = K * Cout * CinP;
        wprep_kernel<<<eb(wtot), 256, 0, stream>>>(W, WT, K, Cin, Cout, CinP, wtot);
        zero(y, (size_t)nmax * Cout);
        for (int k = 0; k < K; k++) {
            int np = rb[k].np; if (!np) continue;
            sp(CinP, Cout, xh, WT + (size_t)k * Cout * CinP, rb[k].pairs, np, y, tr ? 1 : 0);
        }
    };
    // BN + leaky relu -> packed bf16 activations (XH)
    auto bnact = [&](const float* x, BN bn, const int* nd, int C, float leak) {
        hipMemsetAsync(stats, 0, 2 * C * sizeof(float), stream);
        colstats_kernel<<<1024, 256, 2 * C * sizeof(float), stream>>>(x, stats, nd, C);
        bnact_bf16_kernel<<<eb((long long)nmax * C), 256, 0, stream>>>(
            x, XH, stats, bn.g, bn.b, nd, C, leak);
    };
    auto dense = [&](const __bf16* xh, int CinP, const float* W, const float* bias,
                     float* y, const int* nd, int Cin, int Cout) {
        int wtot = Cout * CinP;
        wprep_kernel<<<eb(wtot), 256, 0, stream>>>(W, WT, 1, Cin, Cout, CinP, wtot);
        dn(CinP, Cout, xh, bias, y, nd);
    };
    auto blockrun = [&](const float* x, int Cin, int Cout, float* out,
                        const Blk& p, const RB* rb, const int* nd) {
        const float* sc = x;
        if (p.nin) {   // NiN shortcut: convert x once, dense WMMA
            cvt_bf16_kernel<<<eb((long long)nmax * Cin), 256, 0, stream>>>(
                x, XH, nd, Cin, Cin);
            dense(XH, Cin, p.nin, nullptr, SCb, nd, Cin, Cout);
            sc = SCb;
        }
        bnact(x, p.bn1, nd, Cin, 0.2f);
        conv(XH, Cin, p.w1, rb, 27, Cin, Cout, T2, false);
        bnact(T2, p.bn2, nd, Cout, 0.2f);
        conv(XH, Cout, p.w2, rb, 27, Cout, Cout, out, false);
        add_kernel<<<eb((long long)nmax * Cout), 256, 0, stream>>>(out, sc, nd, Cout);
    };

    // ---- encoder -----------------------------------------------------------
    float* A = G0; float* B = G1;
    // stem conv: Cin=1 padded to stride 8 (keeps gather loads 16B-aligned)
    cvt_bf16_kernel<<<eb((long long)nmax * 8), 256, 0, stream>>>(
        features, XH, ndev[0], 1, 8);
    conv(XH, 8, w_in, rb_sub[0], 27, 1, 16, A, false);
    for (int i = 0; i < 5; i++) {
        int C = NPLh[i];
        for (int j = 0; j < 2; j++) {
            blockrun(A, C, C, B, enc[i][j], rb_sub[i], ndev[i]);
            float* t = A; A = B; B = t;
        }
        if (i < 4) {
            d2d(FM[i], A, (size_t)nmax * C);
            bnact(A, down_bn[i], ndev[i], C, 0.2f);
            conv(XH, C, down_w[i], rb_down[i], 8, C, NPLh[i + 1], B, false);
            float* t = A; A = B; B = t;
        }
    }
    d2d(E4, A, (size_t)nmax * 80);

    // ---- decoder (run twice) ----------------------------------------------
    auto decode = [&](Dec* dps) -> float* {
        d2d(G0, E4, (size_t)nmax * 80);
        float* a = G0; float* b = G1;
        for (int i = 0; i < 4; i++) {
            int l = 3 - i, Cc = NPLh[l], Cup = NPLh[l + 1];
            bnact(a, dps[i].bn, ndev[l + 1], Cup, 0.2f);
            conv(XH, Cup, dps[i].wdec, rb_down[l], 8, Cup, Cc, T2, true);   // deconv
            copycols_kernel<<<eb((long long)nmax * Cc), 256, 0, stream>>>(
                FM[l], Cc, b, 2 * Cc, 0, ndev[l]);
            copycols_kernel<<<eb((long long)nmax * Cc), 256, 0, stream>>>(
                T2, Cc, b, 2 * Cc, Cc, ndev[l]);
            { float* t = a; a = b; b = t; }
            blockrun(a, 2 * Cc, Cc, b, dps[i].b0, rb_sub[l], ndev[l]);
            { float* t = a; a = b; b = t; }
            blockrun(a, Cc, Cc, b, dps[i].b1, rb_sub[l], ndev[l]);
            { float* t = a; a = b; b = t; }
        }
        return a;
    };

    float* x1 = decode(dec1);  d2d(X1, x1, (size_t)nmax * 16);
    float* x2 = decode(dec2);  d2d(X2, x2, (size_t)nmax * 16);

    float* outp = (float*)d_out;
    float* seg  = outp;
    float* emb  = outp + (size_t)n0 * 5;
    float* seed = outp + (size_t)n0 * 21;

    // ---- seg head ----------------------------------------------------------
    bnact(X1, seg_bn, ndev[0], 16, 0.0f);
    dense(XH, 16, seg_w, seg_b, seg, ndev[0], 16, 5);

    // ---- emb head ----------------------------------------------------------
    d2d(G0, X1, (size_t)nmax * 16);
    {
        float* a = G0; float* b = G1;
        for (int i = 0; i < 3; i++) {
            blockrun(a, 16, 16, b, embB[i], rb_sub[0], ndev[0]);
            float* t = a; a = b; b = t;
        }
        bnact(a, emb_bn, ndev[0], 16, 0.0f);
        dense(XH, 16, emb_w, emb_b, emb, ndev[0], 16, 16);
    }

    // ---- seed head ---------------------------------------------------------
    d2d(G0, X2, (size_t)nmax * 16);
    {
        float* a = G0; float* b = G1;
        for (int i = 0; i < 3; i++) {
            blockrun(a, 16, 16, b, seedB[i], rb_sub[0], ndev[0]);
            float* t = a; a = b; b = t;
        }
        bnact(a, seed_bn, ndev[0], 16, 0.0f);
        dense(XH, 16, seed_w, seed_b, seed, ndev[0], 16, 2);
        sigmoid_col0_kernel<<<eb(nmax), 256, 0, stream>>>(seed, 2, ndev[0]);
    }
}
